// edr_loss_23691039605310
// MI455X (gfx1250) — matrix operand order repaired and checked by reading, not compiled, not run on hardware
//
#include <hip/hip_runtime.h>
#include <hip/hip_bf16.h>

// ---------------------------------------------------------------------------
// EDR loss pipeline for MI455X (gfx1250, wave32, WMMA).
//   Stage 1: pack weighted real spectrum into WMMA-B layout (f16)
//   Stage 2: irfft as cosine-DFT GEMM over t < 96016 only (signal is even:
//            x[t] = x[N-t] for a real spectrum)  ->  2x FLOP reduction.
//            B staged in LDS (double-buffered, async-to-LDS when available)
//            and shared by 8 waves x 2 M-tiles -> 16x less L2 traffic.
//   Stage 3: mirror + zero-pad; pack windowed frames + trig tables
//   Stage 4: STFT as cos/sin GEMMs, A (cos/sin) staged in LDS, -> power
//   Stage 5: EDR suffix-scan + deterministic LDS-tree reduction
// ---------------------------------------------------------------------------

typedef __attribute__((ext_vector_type(16))) _Float16 v16h;
typedef __attribute__((ext_vector_type(8)))  float    v8f;
typedef __attribute__((ext_vector_type(4)))  int      v4i;

constexpr int BATCH = 32;
constexpr int NFREQ = 192000;     // input row length
constexpr int NIR   = 192000;     // irfft length
constexpr int NH    = 96001;      // one-sided bins actually used
constexpr int KT_IR = 3001;       // 96032 / 32 K-tiles for irfft GEMM
constexpr int NSIG  = 64;         // 32 target + 32 achieved signals
constexpr int LP    = 192512;     // padded signal length (pad=512)
constexpr int WIN   = 4096;
constexpr int HOP   = 2048;
constexpr int TFR   = 93;         // frames
constexpr int FB    = 2049;       // rfft bins
constexpr int MT_S  = 129;        // freq tiles (2064 rows, 2049 valid)
constexpr int NT_S  = 372;        // 64*93 / 16 column tiles (exact)
constexpr int KT_S  = 128;        // 4096 / 32
constexpr int NCOL  = NSIG * TFR; // 5952
constexpr int MT_C  = 6016;       // computed time tiles (t < 96256)
constexpr int TMIR  = 96016;      // mirror region start

// workspace layout (bytes, all 256-aligned by construction)
constexpr size_t SZ_BSPEC = (size_t)KT_IR * 4 * 512 * 2;           // 12,292,096
constexpr size_t SZ_X     = (size_t)NSIG * LP * 4;                 // 49,283,072
constexpr size_t SZ_TRIG  = (size_t)MT_S * KT_S * 512 * 2;         // 16,908,288
constexpr size_t SZ_BF    = (size_t)KT_S * NT_S * 512 * 2 + 16384; // + slack
constexpr size_t SZ_P     = (size_t)FB * NCOL * 4 + 16384;         // + slack
constexpr size_t OFF_BSPEC = 0;
constexpr size_t OFF_X     = OFF_BSPEC + SZ_BSPEC;
constexpr size_t OFF_ACOS  = OFF_X + SZ_X;
constexpr size_t OFF_ASIN  = OFF_ACOS + SZ_TRIG;
constexpr size_t OFF_BF    = OFF_ASIN + SZ_TRIG;
constexpr size_t OFF_P     = OFF_BF + SZ_BF;
constexpr size_t OFF_ND    = OFF_P + SZ_P;                         // 64 floats

__device__ __forceinline__ v8f wmma16(v16h a, v16h b, v8f c) {
  // (neg_a, A, neg_b, B, c_mod, C, reuse_a, reuse_b)
  return __builtin_amdgcn_wmma_f32_16x16x32_f16(false, a, false, b,
                                                (short)0, c, false, false);
}

// ---- async global->LDS copy (16B per thread), guarded fallback -------------
#if defined(__has_builtin)
#if __has_builtin(__builtin_amdgcn_global_load_async_to_lds_b128)
#define EDR_ASYNC_LDS 1
#endif
#endif

__device__ __forceinline__ void copy16(const void* g, void* l) {
#ifdef EDR_ASYNC_LDS
  // signature (from diagnostic): (global v4i*, local v4i*, imm offset, imm cpol)
  __builtin_amdgcn_global_load_async_to_lds_b128(
      (__attribute__((address_space(1))) v4i*)(g),
      (__attribute__((address_space(3))) v4i*)(l), 0, 0);
#else
  *(float4*)l = *(const float4*)g;
#endif
}

__device__ __forceinline__ void async_wait0() {
#ifdef EDR_ASYNC_LDS
#if __has_builtin(__builtin_amdgcn_s_wait_asynccnt)
  __builtin_amdgcn_s_wait_asynccnt(0);
#endif
#endif
}

// ---- Stage 1: weighted spectrum -> WMMA B layout ---------------------------
// B tile (32x16 f16): lane l: N=l&15; half h -> K = (h<8?h:h+8) + (l<16?0:8)
__global__ void k_pack_spec(const float* __restrict__ tgt,
                            const float* __restrict__ ach,
                            _Float16* __restrict__ P) {
  int idx = blockIdx.x * 256 + threadIdx.x;
  if (idx >= KT_IR * 4 * 512) return;
  int h  = idx & 15;
  int l  = (idx >> 4) & 31;
  int nt = (idx >> 9) & 3;
  int kt = idx >> 11;
  int k   = kt * 32 + ((h < 8) ? h : h + 8) + ((l < 16) ? 0 : 8);
  int sig = nt * 16 + (l & 15);
  float v = 0.0f;
  if (k < NH) {
    const float* src = (sig < BATCH) ? tgt : ach;
    float xv = src[(size_t)(sig & (BATCH - 1)) * NFREQ + k];
    v = xv * ((k == 0 || k == NH - 1) ? 1.0f : 2.0f);  // 1/N applied later
  }
  P[idx] = (_Float16)v;
}

// ---- Stage 3a: cos/sin DFT tables in WMMA A layout -------------------------
// A tile (16x32 f16): lane l: M=l&15; half h -> K = (h<8?h:h+8) + (l<16?0:8)
__global__ void k_trig(_Float16* __restrict__ Ac, _Float16* __restrict__ As) {
  int idx = blockIdx.x * 256 + threadIdx.x;
  if (idx >= MT_S * KT_S * 512) return;
  int h  = idx & 15;
  int l  = (idx >> 4) & 31;
  int kt = (idx >> 9) & 127;
  int mt = idx >> 16;
  int f    = mt * 16 + (l & 15);
  int koff = ((h < 8) ? h : h + 8) + ((l < 16) ? 0 : 8);
  int n    = kt * 32 + koff;
  float ang = 6.28318530718f * (float)((f * n) & (WIN - 1)) * (1.0f / WIN);
  Ac[idx] = (_Float16)__cosf(ang);
  As[idx] = (_Float16)__sinf(ang);
}

// ---- Stage 2: irfft cosine-DFT GEMM with LDS-staged B ----------------------
// WG = 8 waves; wave w owns M-tiles {blk*16+2w, +1}; all waves share the
// same 4KB B chunk per K-step, staged via async copy, double buffered.
__global__ void __launch_bounds__(256) k_irfft(const _Float16* __restrict__ Bs,
                                               float* __restrict__ x) {
  __shared__ _Float16 lds[4096];  // 2 x 4KB
  int tid = threadIdx.x;
  int wave = tid >> 5, lane = tid & 31;
  int m0 = blockIdx.x * 16 + wave * 2;  // first of two M-tiles
  int t0 = m0 * 16 + (lane & 15);
  int t1 = t0 + 16;
  int s0 = (32 * t0) % NIR;
  int s1 = (32 * t1) % NIR;
  int kbase = (lane < 16) ? 0 : 8;
  int p0[16], p1[16];
#pragma unroll
  for (int h = 0; h < 16; ++h) {
    int koff = ((h < 8) ? h : h + 8) + kbase;
    p0[h] = (koff * t0) % NIR;
    p1[h] = (koff * t1) % NIR;
  }
  v8f c00 = {}, c01 = {}, c02 = {}, c03 = {};
  v8f c10 = {}, c11 = {}, c12 = {}, c13 = {};
  const float c1 = 6.28318530717958647692f / (float)NIR;

  // prologue: stage K-tile 0 into buffer 0
  copy16((const char*)Bs + (tid << 4), (char*)lds + (tid << 4));

  for (int kt = 0; kt < KT_IR; ++kt) {
    async_wait0();
    __syncthreads();                     // chunk for buf (kt&1) visible
    if (kt + 1 < KT_IR) {
      int nb = (kt + 1) & 1;
      copy16((const char*)Bs + ((size_t)(kt + 1) << 12) + (tid << 4),
             (char*)lds + ((size_t)nb << 12) + (tid << 4));
    }
    v16h a0, a1;
#pragma unroll
    for (int h = 0; h < 16; ++h) {
      a0[h] = (_Float16)__cosf((float)p0[h] * c1);
      a1[h] = (_Float16)__cosf((float)p1[h] * c1);
      p0[h] += s0; if (p0[h] >= NIR) p0[h] -= NIR;
      p1[h] += s1; if (p1[h] >= NIR) p1[h] -= NIR;
    }
    const _Float16* lb = lds + ((kt & 1) << 11) + lane * 16;
    v16h b0 = *(const v16h*)(lb);
    v16h b1 = *(const v16h*)(lb + 512);
    v16h b2 = *(const v16h*)(lb + 1024);
    v16h b3 = *(const v16h*)(lb + 1536);
    c00 = wmma16(a0, b0, c00);  c10 = wmma16(a1, b0, c10);
    c01 = wmma16(a0, b1, c01);  c11 = wmma16(a1, b1, c11);
    c02 = wmma16(a0, b2, c02);  c12 = wmma16(a1, b2, c12);
    c03 = wmma16(a0, b3, c03);  c13 = wmma16(a1, b3, c13);
    __syncthreads();                     // reads done before re-stage
  }
  // C layout: VGPR r, lane<16 -> (M=r, N=lane); lane>=16 -> (M=r+8, N=lane-16)
  float inv = 1.0f / (float)NIR;
  int tb0 = m0 * 16 + ((lane < 16) ? 0 : 8);
  int tb1 = tb0 + 16;
  int sc = lane & 15;
#pragma unroll
  for (int r = 0; r < 8; ++r) {
    x[(size_t)(sc)*LP      + tb0 + r] = c00[r] * inv;
    x[(size_t)(16 + sc)*LP + tb0 + r] = c01[r] * inv;
    x[(size_t)(32 + sc)*LP + tb0 + r] = c02[r] * inv;
    x[(size_t)(48 + sc)*LP + tb0 + r] = c03[r] * inv;
    x[(size_t)(sc)*LP      + tb1 + r] = c10[r] * inv;
    x[(size_t)(16 + sc)*LP + tb1 + r] = c11[r] * inv;
    x[(size_t)(32 + sc)*LP + tb1 + r] = c12[r] * inv;
    x[(size_t)(48 + sc)*LP + tb1 + r] = c13[r] * inv;
  }
}

// ---- Stage 3: mirror (x[t] = x[N-t]) and zero tail -------------------------
__global__ void k_mirror(float* __restrict__ x) {
  int idx = blockIdx.x * 256 + threadIdx.x;
  if (idx >= NSIG * (NIR - TMIR)) return;
  int sig = idx / (NIR - TMIR);
  int t   = TMIR + idx % (NIR - TMIR);
  x[(size_t)sig * LP + t] = x[(size_t)sig * LP + (NIR - t)];
}

__global__ void k_ztail(float* __restrict__ x) {
  int idx = blockIdx.x * 256 + threadIdx.x;  // 64 * 512 pad samples
  if (idx >= NSIG * (LP - NIR)) return;
  int sig = idx >> 9, o = idx & 511;
  x[(size_t)sig * LP + NIR + o] = 0.0f;
}

// ---- Stage 3b: windowed frames -> WMMA B layout ----------------------------
__global__ void k_pack_frames(const float* __restrict__ x,
                              _Float16* __restrict__ Bf) {
  int idx = blockIdx.x * 256 + threadIdx.x;
  if (idx >= KT_S * NT_S * 512) return;
  int h   = idx & 15;
  int l   = (idx >> 4) & 31;
  int rem = idx >> 9;
  int nt  = rem % NT_S;
  int kt  = rem / NT_S;
  int col = nt * 16 + (l & 15);
  int sig = col / TFR, tf = col % TFR;
  int n   = kt * 32 + ((h < 8) ? h : h + 8) + ((l < 16) ? 0 : 8);
  float w = 0.5f - 0.5f * __cosf(6.28318530718f * (float)n * (1.0f / WIN));
  float v = w * x[(size_t)sig * LP + (size_t)tf * HOP + n];
  Bf[idx] = (_Float16)v;
}

// ---- Stage 4: STFT GEMM, LDS-staged cos/sin A, -> power --------------------
// WG = 8 waves: 2 M-tiles (shared, staged) x 8 column pairs (one per wave).
__global__ void __launch_bounds__(256) k_stft(const _Float16* __restrict__ Ac,
                                              const _Float16* __restrict__ As,
                                              const _Float16* __restrict__ Bf,
                                              float* __restrict__ P) {
  __shared__ _Float16 lds[4096];  // 2 x (4 chunks x 1KB)
  int tid = threadIdx.x;
  int wave = tid >> 5, lane = tid & 31;
  int mtg = blockIdx.x / 24;
  int ntg = blockIdx.x % 24;
  int mt0 = mtg * 2;
  int mt1 = (mt0 + 1 < MT_S) ? mt0 + 1 : MT_S - 1;
  int pair = ntg * 8 + wave;
  int nt0 = pair * 2, nt1 = nt0 + 1;

  int cch = tid >> 6;           // chunk 0..3 : mt0cos, mt0sin, mt1cos, mt1sin
  int wi  = tid & 63;           // 16B unit within chunk
  int mts = (cch < 2) ? mt0 : mt1;
  const _Float16* tbl = (cch & 1) ? As : Ac;
  const char* gbase = (const char*)(tbl + ((size_t)mts * KT_S) * 512) + (wi << 4);
  char* l0 = (char*)lds + (cch << 10) + (wi << 4);

  v8f re00 = {}, im00 = {}, re01 = {}, im01 = {};
  v8f re10 = {}, im10 = {}, re11 = {}, im11 = {};

  copy16(gbase, l0);            // stage kt=0 into buffer 0
  for (int kt = 0; kt < KT_S; ++kt) {
    async_wait0();
    __syncthreads();
    if (kt + 1 < KT_S)
      copy16(gbase + ((size_t)(kt + 1) << 10),
             l0 + ((size_t)((kt + 1) & 1) << 12));
    const _Float16* lb = lds + ((kt & 1) << 11) + lane * 16;
    v16h ac0 = *(const v16h*)(lb);
    v16h as0 = *(const v16h*)(lb + 512);
    v16h ac1 = *(const v16h*)(lb + 1024);
    v16h as1 = *(const v16h*)(lb + 1536);
    const _Float16* bp = Bf + ((size_t)kt * NT_S + nt0) * 512 + lane * 16;
    v16h b0 = *(const v16h*)(bp);
    v16h b1 = *(const v16h*)(bp + 512);
    re00 = wmma16(ac0, b0, re00);  im00 = wmma16(as0, b0, im00);
    re01 = wmma16(ac0, b1, re01);  im01 = wmma16(as0, b1, im01);
    re10 = wmma16(ac1, b0, re10);  im10 = wmma16(as1, b0, im10);
    re11 = wmma16(ac1, b1, re11);  im11 = wmma16(as1, b1, im11);
    __syncthreads();
  }
  int fb0 = mt0 * 16 + ((lane < 16) ? 0 : 8);
  int fb1 = mt1 * 16 + ((lane < 16) ? 0 : 8);
  int c0 = nt0 * 16 + (lane & 15);
  int c1 = c0 + 16;
#pragma unroll
  for (int r = 0; r < 8; ++r) {
    int f0 = fb0 + r, f1 = fb1 + r;
    if (f0 < FB && c0 < NCOL)
      P[(size_t)f0 * NCOL + c0] = re00[r] * re00[r] + im00[r] * im00[r];
    if (f0 < FB && c1 < NCOL)
      P[(size_t)f0 * NCOL + c1] = re01[r] * re01[r] + im01[r] * im01[r];
    if (f1 < FB && c0 < NCOL)
      P[(size_t)f1 * NCOL + c0] = re10[r] * re10[r] + im10[r] * im10[r];
    if (f1 < FB && c1 < NCOL)
      P[(size_t)f1 * NCOL + c1] = re11[r] * re11[r] + im11[r] * im11[r];
  }
}

// ---- Stage 5: EDR suffix-scan + deterministic reduction --------------------
__global__ void __launch_bounds__(256) k_edr(const float* __restrict__ P,
                                             float* __restrict__ nd) {
  __shared__ float sn[256], sd[256];
  int b = blockIdx.x;  // 0..31
  float num = 0.0f, den = 0.0f;
  for (int f = threadIdx.x; f < FB; f += 256) {
    const float* pt = P + (size_t)f * NCOL + b * TFR;
    const float* pa = P + (size_t)f * NCOL + (b + BATCH) * TFR;
    float st = 0.0f, sa = 0.0f;
    for (int t = TFR - 1; t >= 0; --t) {
      st += pt[t];
      sa += pa[t];
      float et = 10.0f * __log10f(st);
      float ea = 10.0f * __log10f(sa);
      num += fabsf(et - ea);
      den += fabsf(et);
    }
  }
  sn[threadIdx.x] = num;
  sd[threadIdx.x] = den;
  __syncthreads();
  for (int s = 128; s > 0; s >>= 1) {
    if (threadIdx.x < s) {
      sn[threadIdx.x] += sn[threadIdx.x + s];
      sd[threadIdx.x] += sd[threadIdx.x + s];
    }
    __syncthreads();
  }
  if (threadIdx.x == 0) {
    nd[b] = sn[0];
    nd[BATCH + b] = sd[0];
  }
}

__global__ void k_final(const float* __restrict__ nd, float* __restrict__ out) {
  int l = threadIdx.x;  // one wave32
  float r = nd[l] / nd[BATCH + l];
  for (int off = 16; off > 0; off >>= 1) r += __shfl_down(r, off, 32);
  if (l == 0) out[0] = r;
}

// ---------------------------------------------------------------------------
extern "C" void kernel_launch(void* const* d_in, const int* in_sizes, int n_in,
                              void* d_out, int out_size, void* d_ws,
                              size_t ws_size, hipStream_t stream) {
  const float* tgt = (const float*)d_in[0];
  const float* ach = (const float*)d_in[1];
  float* out = (float*)d_out;
  char* ws = (char*)d_ws;

  _Float16* Bs = (_Float16*)(ws + OFF_BSPEC);
  float*    x  = (float*)   (ws + OFF_X);
  _Float16* Ac = (_Float16*)(ws + OFF_ACOS);
  _Float16* As = (_Float16*)(ws + OFF_ASIN);
  _Float16* Bf = (_Float16*)(ws + OFF_BF);
  float*    P  = (float*)   (ws + OFF_P);
  float*    nd = (float*)   (ws + OFF_ND);

  k_pack_spec<<<(KT_IR * 4 * 512 + 255) / 256, 256, 0, stream>>>(tgt, ach, Bs);
  k_trig<<<(MT_S * KT_S * 512 + 255) / 256, 256, 0, stream>>>(Ac, As);
  k_irfft<<<MT_C / 16, 256, 0, stream>>>(Bs, x);
  k_mirror<<<(NSIG * (NIR - TMIR) + 255) / 256, 256, 0, stream>>>(x);
  k_ztail<<<(NSIG * (LP - NIR) + 255) / 256, 256, 0, stream>>>(x);
  k_pack_frames<<<(KT_S * NT_S * 512 + 255) / 256, 256, 0, stream>>>(x, Bf);
  k_stft<<<65 * 24, 256, 0, stream>>>(Ac, As, Bf, P);
  k_edr<<<BATCH, 256, 0, stream>>>(P, nd);
  k_final<<<1, 32, 0, stream>>>(nd, out);
}
